// Qwen3NextGatedDeltaNet_46145128628426
// MI455X (gfx1250) — matrix operand order repaired and checked by reading, not compiled, not run on hardware
//
#include <hip/hip_runtime.h>
#include <stdint.h>

// ---------------- types for CDNA5 WMMA ----------------
typedef __attribute__((ext_vector_type(16))) __bf16 v16bf;
typedef __attribute__((ext_vector_type(8)))  float  v8f;

union BF8 { uint4 u; __bf16 h[8]; };   // 16 bytes == 8 bf16

// ---------------- CDNA5 async-to-LDS availability probe ----------------
#if defined(__has_builtin)
#if __has_builtin(__builtin_amdgcn_global_load_async_to_lds_b32) && \
    __has_builtin(__builtin_amdgcn_s_wait_asynccnt)
#define HAVE_ASYNC_LDS 1
#endif
#endif

#ifdef HAVE_ASYNC_LDS
__device__ __forceinline__ void async_ld_f32(const float* g, float* l) {
    __builtin_amdgcn_global_load_async_to_lds_b32(
        (__attribute__((address_space(1))) int*)(g),
        (__attribute__((address_space(3))) int*)(l), 0, 0);
}
#endif

// ---------------- problem constants ----------------
constexpr int L_SEQ   = 2048;
constexpr int H_DIM   = 2048;
constexpr int HV      = 32;
constexpr int HK      = 16;
constexpr int DK      = 128;
constexpr int DV      = 128;
constexpr int KEY_DIM   = HK * DK;        // 2048
constexpr int VALUE_DIM = HV * DV;        // 4096
constexpr int PROJ_N  = 2 * KEY_DIM + 2 * VALUE_DIM;  // 12288
constexpr int CONV_DIM = 2 * KEY_DIM + VALUE_DIM;     // 8192

// ---------------- workspace layout (bytes) ----------------
constexpr size_t OFF_HBF    = 0;            // 8,388,608   (phase 1)
constexpr size_t OFF_WQKVZS = 8388608;      // 50,331,648  (phase 1)
constexpr size_t OFF_COREBF = 0;            // 16,777,216  (phase 2, aliases HBF)
constexpr size_t OFF_CORE   = 16777216;     // 33,554,432  (phase 2, aliases WQKVZS)
constexpr size_t OFF_WBAS   = 58720256;     //    262,144
constexpr size_t OFF_WOUTS  = 58982400;     // 16,777,216
constexpr size_t OFF_PROJ   = 75759616;     // 100,663,296
constexpr size_t OFF_BA     = 176422912;    //    524,288
constexpr size_t OFF_QBUF   = 176947200;    // 16,777,216
constexpr size_t OFF_KBUF   = 193724416;    // 16,777,216
constexpr size_t OFF_VBUF   = 210501632;    // 33,554,432
constexpr size_t OFF_GBUF   = 244056064;    //    262,144
constexpr size_t OFF_BETA   = 244318208;    //    262,144

// ---------------- helpers ----------------
__device__ __forceinline__ uint16_t f32_to_bf16_rne(float f) {
    uint32_t x = __float_as_uint(f);
    x += 0x7FFFu + ((x >> 16) & 1u);
    return (uint16_t)(x >> 16);
}
__device__ __forceinline__ float silu(float x) { return x / (1.f + expf(-x)); }

// ---------------- f32 -> bf16 (row-major, for A matrices) ----------------
__global__ void f32_to_bf16_kernel(const float* __restrict__ in,
                                   uint16_t* __restrict__ out, int n) {
    int i = blockIdx.x * blockDim.x + threadIdx.x;
    if (i < n) out[i] = f32_to_bf16_rne(in[i]);
}

// ---------------- f32 -> bf16 with WMMA-B swizzle ----------------
// out[ ((k/32)*N + n)*32 + (k%32) ]: a lane's B fragment (K = kb+16*half+e,
// fixed column n) becomes 16 contiguous bf16 = two 16B loads.
__global__ void convert_swizzleB_kernel(const float* __restrict__ W,
                                        uint16_t* __restrict__ out, int K, int N) {
    size_t i = (size_t)blockIdx.x * blockDim.x + threadIdx.x;
    if (i >= (size_t)K * N) return;
    int k = (int)(i / N), n = (int)(i % N);
    out[((size_t)(k >> 5) * N + n) * 32 + (k & 31)] = f32_to_bf16_rne(W[i]);
}

// ---------------- bf16 WMMA GEMM: C[M,N] = A[M,K] * B[K,N] ----------------
// Branch-free (NT subtiles exactly tile N), 2x-unrolled ping-pong pipeline:
// loads for k-block i+1 are issued before the WMMAs of k-block i, with the
// two pipeline stages living in disjoint SSA values (no register copies,
// no WMMA->VALU hazard nops). Requires K % 64 == 0 (true for all calls).
template <int NT>
__global__ __launch_bounds__(128)
void wmma_gemm_kernel(const uint16_t* __restrict__ A, const uint16_t* __restrict__ Bsw,
                      float* __restrict__ C, int M, int N, int K) {
    const int lane = threadIdx.x & 31;
    const int wave = threadIdx.x >> 5;
    const int half = lane >> 4;        // which K sub-range this lane holds
    const int l    = lane & 15;        // row (A) / column (B,C) within tile
    const int m0   = (blockIdx.y * 4 + wave) * 16;
    const int n0   = blockIdx.x * (NT * 16);

    const uint16_t* arow = A + (size_t)(m0 + l) * K + 16 * half;
    const uint16_t* bcol = Bsw + (size_t)(n0 + l) * 32 + 16 * half;
    const size_t bstride = (size_t)N * 32;           // elements per 32-k block

    auto load_a = [&](const uint16_t* p) {
        BF8 lo, hi;
        lo.u = *(const uint4*)(p);
        hi.u = *(const uint4*)(p + 8);
        v16bf f;
        #pragma unroll
        for (int e = 0; e < 8; ++e) { f[e] = lo.h[e]; f[8 + e] = hi.h[e]; }
        return f;
    };
    auto load_b = [&](const uint16_t* p, int nt) {
        const uint16_t* bp = p + nt * (16 * 32);
        BF8 lo, hi;
        lo.u = *(const uint4*)bp;
        hi.u = *(const uint4*)(bp + 8);
        v16bf f;
        #pragma unroll
        for (int e = 0; e < 8; ++e) { f[e] = lo.h[e]; f[8 + e] = hi.h[e]; }
        return f;
    };

    v8f acc[NT] = {};

    // stage 0 preload
    v16bf a0 = load_a(arow);
    v16bf b0[NT];
    #pragma unroll
    for (int nt = 0; nt < NT; ++nt) b0[nt] = load_b(bcol, nt);

    for (int kb = 0; kb < K; kb += 64) {
        // load stage 1 (kb+32, always in range since K % 64 == 0)
        const uint16_t* ap1 = arow + kb + 32;
        const uint16_t* bp1 = bcol + ((size_t)(kb >> 5) + 1) * bstride;
        v16bf a1 = load_a(ap1);
        v16bf b1[NT];
        #pragma unroll
        for (int nt = 0; nt < NT; ++nt) b1[nt] = load_b(bp1, nt);

        // compute stage 0
        #pragma unroll
        for (int nt = 0; nt < NT; ++nt)
            acc[nt] = __builtin_amdgcn_wmma_f32_16x16x32_bf16(
                false, a0, false, b0[nt], (short)0, acc[nt], false, false);

        // load stage 0 for next trip (wrap to block 0 on last trip: branch-free)
        const bool wrap = (kb + 64 >= K);
        const uint16_t* ap2 = wrap ? arow : arow + kb + 64;
        const uint16_t* bp2 = wrap ? bcol : bcol + ((size_t)(kb >> 5) + 2) * bstride;
        a0 = load_a(ap2);
        #pragma unroll
        for (int nt = 0; nt < NT; ++nt) b0[nt] = load_b(bp2, nt);

        // compute stage 1
        #pragma unroll
        for (int nt = 0; nt < NT; ++nt)
            acc[nt] = __builtin_amdgcn_wmma_f32_16x16x32_bf16(
                false, a1, false, b1[nt], (short)0, acc[nt], false, false);
    }

    // C/D layout: lane l = column, VGPR r = row (r + 8*half)
    float* crow = C + (size_t)(m0 + 8 * half) * N + n0 + l;
    #pragma unroll
    for (int nt = 0; nt < NT; ++nt) {
        #pragma unroll
        for (int r = 0; r < 8; ++r) crow[(size_t)r * N + nt * 16] = acc[nt][r];
    }
}

// ---------------- conv + SiLU + per-head L2-norm for q/k ----------------
__global__ __launch_bounds__(128)
void conv_qk_kernel(const float* __restrict__ proj, const float* __restrict__ conv_w,
                    float* __restrict__ qbuf, float* __restrict__ kbuf) {
    const int t = blockIdx.x, head = blockIdx.y, d = threadIdx.x;
    const int c = head * 128 + d;                 // channel in [0, 4096)
    float acc = 0.f;
    #pragma unroll
    for (int i = 0; i < 4; ++i) {
        int tt = t - 3 + i;
        if (tt >= 0) acc += proj[(size_t)tt * PROJ_N + c] * conv_w[c * 4 + i];
    }
    float s = silu(acc);
    __shared__ float red[128];
    red[d] = s * s;
    __syncthreads();
    for (int off = 64; off > 0; off >>= 1) {
        if (d < off) red[d] += red[d + off];
        __syncthreads();
    }
    float inv = rsqrtf(red[0] + 1e-6f);
    if (head < HK) qbuf[((size_t)t * HK + head) * 128 + d] = s * inv * 0.08838834764831845f;
    else           kbuf[((size_t)t * HK + (head - HK)) * 128 + d] = s * inv;
}

// ---------------- conv + SiLU for v ----------------
__global__ void conv_v_kernel(const float* __restrict__ proj, const float* __restrict__ conv_w,
                              float* __restrict__ vbuf) {
    size_t gid = (size_t)blockIdx.x * blockDim.x + threadIdx.x;
    if (gid >= (size_t)L_SEQ * VALUE_DIM) return;
    int t = (int)(gid >> 12);
    int c = 2 * KEY_DIM + (int)(gid & 4095);
    float acc = 0.f;
    #pragma unroll
    for (int i = 0; i < 4; ++i) {
        int tt = t - 3 + i;
        if (tt >= 0) acc += proj[(size_t)tt * PROJ_N + c] * conv_w[c * 4 + i];
    }
    vbuf[gid] = silu(acc);
}

// ---------------- beta / g gates ----------------
__global__ void gates_kernel(const float* __restrict__ ba, const float* __restrict__ dt_bias,
                             const float* __restrict__ A_log,
                             float* __restrict__ gbuf, float* __restrict__ betabuf) {
    int gid = blockIdx.x * blockDim.x + threadIdx.x;
    if (gid >= L_SEQ * HV) return;
    int t = gid >> 5, h = gid & 31;
    float b = ba[t * 64 + h];
    float a = ba[t * 64 + 32 + h];
    float x = a + dt_bias[h];
    float sp = (x > 20.f) ? x : log1pf(expf(x));
    gbuf[gid]    = -expf(A_log[h]) * sp;
    betabuf[gid] = 1.f / (1.f + expf(-b));
}

// ---------------- gated delta scan ----------------
// DV columns of S are independent -> grid (8 dv-chunks, 32 heads) = 256 blocks.
// S[128,16] LDS-resident (stride 17 = conflict-free). k/q/v/g/beta for step
// t+1 are staged into the alternate LDS buffer with CDNA5 async-to-LDS loads
// while step t computes; one s_wait_asynccnt + barrier at the step boundary.
__global__ __launch_bounds__(128)
void scan_kernel(const float* __restrict__ qbuf, const float* __restrict__ kbuf,
                 const float* __restrict__ vbuf, const float* __restrict__ gbuf,
                 const float* __restrict__ betabuf, float* __restrict__ core) {
    const int chunk = blockIdx.x;          // 0..7
    const int h     = blockIdx.y;          // 0..31
    const int hs    = h >> 1;              // source head for q/k (repeat=2)
    const int tid   = threadIdx.x;
    const int g8    = tid >> 4;            // dk group 0..7
    const int dvl   = tid & 15;

    __shared__ float S[128 * 17];
    __shared__ float ks[2][128], qs[2][128], vs[2][16], scal[2][2];
    __shared__ float part[128];

    for (int i = tid; i < 128 * 17; i += 128) S[i] = 0.f;

#ifdef HAVE_ASYNC_LDS
    async_ld_f32(kbuf + (size_t)hs * 128 + tid, &ks[0][tid]);
    async_ld_f32(qbuf + (size_t)hs * 128 + tid, &qs[0][tid]);
    if (tid < 16) async_ld_f32(vbuf + (size_t)h * 128 + chunk * 16 + tid, &vs[0][tid]);
    if (tid == 0) {
        async_ld_f32(gbuf + h, &scal[0][0]);
        async_ld_f32(betabuf + h, &scal[0][1]);
    }
    __builtin_amdgcn_s_wait_asynccnt(0);
#endif
    __syncthreads();

    for (int t = 0; t < L_SEQ; ++t) {
#ifdef HAVE_ASYNC_LDS
        const int buf = t & 1;
        const int nb  = buf ^ 1;
        const int tn  = (t + 1 < L_SEQ) ? t + 1 : t;   // branch-free tail restage
        async_ld_f32(kbuf + ((size_t)tn * HK + hs) * 128 + tid, &ks[nb][tid]);
        async_ld_f32(qbuf + ((size_t)tn * HK + hs) * 128 + tid, &qs[nb][tid]);
        if (tid < 16) async_ld_f32(vbuf + ((size_t)tn * HV + h) * 128 + chunk * 16 + tid, &vs[nb][tid]);
        if (tid == 0) {
            async_ld_f32(gbuf + tn * HV + h, &scal[nb][0]);
            async_ld_f32(betabuf + tn * HV + h, &scal[nb][1]);
        }
#else
        const int buf = 0;
        ks[0][tid] = kbuf[((size_t)t * HK + hs) * 128 + tid];
        qs[0][tid] = qbuf[((size_t)t * HK + hs) * 128 + tid];
        if (tid < 16) vs[0][tid] = vbuf[((size_t)t * HV + h) * 128 + chunk * 16 + tid];
        if (tid == 0) { scal[0][0] = gbuf[t * HV + h]; scal[0][1] = betabuf[t * HV + h]; }
        __syncthreads();
#endif
        const float eg   = expf(scal[buf][0]);
        const float beta = scal[buf][1];

        // pass 1: decay + kv partial
        float p = 0.f;
        #pragma unroll
        for (int j = 0; j < 16; ++j) {
            int dk = g8 * 16 + j;
            float sv = S[dk * 17 + dvl] * eg;
            S[dk * 17 + dvl] = sv;
            p += sv * ks[buf][dk];
        }
        part[g8 * 16 + dvl] = p;
        __syncthreads();

        // every lane redundantly reduces its column's kv (saves a barrier)
        float kv = 0.f;
        #pragma unroll
        for (int g = 0; g < 8; ++g) kv += part[g * 16 + dvl];
        const float dlt = (vs[buf][dvl] - kv) * beta;
        __syncthreads();   // part reuse below

        // pass 2: rank-1 update + output partial
        float o = 0.f;
        #pragma unroll
        for (int j = 0; j < 16; ++j) {
            int dk = g8 * 16 + j;
            float sv = S[dk * 17 + dvl] + ks[buf][dk] * dlt;
            S[dk * 17 + dvl] = sv;
            o += sv * qs[buf][dk];
        }
        part[g8 * 16 + dvl] = o;
        __syncthreads();
        if (tid < 16) {
            float ov = 0.f;
            #pragma unroll
            for (int g = 0; g < 8; ++g) ov += part[g * 16 + tid];
            core[((size_t)t * HV + h) * 128 + chunk * 16 + tid] = ov;
        }
#ifdef HAVE_ASYNC_LDS
        __builtin_amdgcn_s_wait_asynccnt(0);
#endif
        __syncthreads();
    }
}

// ---------------- RMSNorm * silu(z) gating, write bf16 ----------------
__global__ __launch_bounds__(128)
void gate_norm_kernel(const float* __restrict__ core, const float* __restrict__ proj,
                      const float* __restrict__ norm_w, uint16_t* __restrict__ corebf) {
    const int t = blockIdx.x, h = blockIdx.y, d = threadIdx.x;
    float x = core[((size_t)t * HV + h) * 128 + d];
    __shared__ float red[128];
    red[d] = x * x;
    __syncthreads();
    for (int off = 64; off > 0; off >>= 1) {
        if (d < off) red[d] += red[d + off];
        __syncthreads();
    }
    float inv = rsqrtf(red[0] * (1.f / 128.f) + 1e-6f);
    float z = proj[(size_t)t * PROJ_N + CONV_DIM + h * 128 + d];
    float val = x * inv * norm_w[d] * silu(z);
    corebf[((size_t)t * HV + h) * 128 + d] = f32_to_bf16_rne(val);
}

// ---------------- launcher ----------------
extern "C" void kernel_launch(void* const* d_in, const int* in_sizes, int n_in,
                              void* d_out, int out_size, void* d_ws, size_t ws_size,
                              hipStream_t stream) {
    const float* hidden  = (const float*)d_in[0];
    const float* Wqkvz   = (const float*)d_in[1];
    const float* Wba     = (const float*)d_in[2];
    const float* conv_w  = (const float*)d_in[3];
    const float* dt_bias = (const float*)d_in[4];
    const float* A_log   = (const float*)d_in[5];
    const float* norm_w  = (const float*)d_in[6];
    const float* Wout    = (const float*)d_in[7];
    float* out = (float*)d_out;

    char* ws = (char*)d_ws;
    uint16_t* hbf     = (uint16_t*)(ws + OFF_HBF);
    uint16_t* wqkvzs  = (uint16_t*)(ws + OFF_WQKVZS);
    uint16_t* wbas    = (uint16_t*)(ws + OFF_WBAS);
    uint16_t* wouts   = (uint16_t*)(ws + OFF_WOUTS);
    uint16_t* corebf  = (uint16_t*)(ws + OFF_COREBF);
    float*    proj    = (float*)(ws + OFF_PROJ);
    float*    ba      = (float*)(ws + OFF_BA);
    float*    qbuf    = (float*)(ws + OFF_QBUF);
    float*    kbuf    = (float*)(ws + OFF_KBUF);
    float*    vbuf    = (float*)(ws + OFF_VBUF);
    float*    gbuf    = (float*)(ws + OFF_GBUF);
    float*    betabuf = (float*)(ws + OFF_BETA);
    float*    corebuf = (float*)(ws + OFF_CORE);

    // precision conversions (~250 MB one-pass => ~11us @ 23.3 TB/s)
    f32_to_bf16_kernel<<<(H_DIM * L_SEQ + 255) / 256, 256, 0, stream>>>(hidden, hbf, H_DIM * L_SEQ);
    convert_swizzleB_kernel<<<(H_DIM * PROJ_N + 255) / 256, 256, 0, stream>>>(Wqkvz, wqkvzs, H_DIM, PROJ_N);
    convert_swizzleB_kernel<<<(H_DIM * 64 + 255) / 256, 256, 0, stream>>>(Wba, wbas, H_DIM, 64);
    convert_swizzleB_kernel<<<(VALUE_DIM * H_DIM + 255) / 256, 256, 0, stream>>>(Wout, wouts, VALUE_DIM, H_DIM);

    // compute-bound GEMMs via v_wmma_f32_16x16x32_bf16
    wmma_gemm_kernel<8><<<dim3(PROJ_N / 128, L_SEQ / 64), 128, 0, stream>>>(hbf, wqkvzs, proj, L_SEQ, PROJ_N, H_DIM);
    wmma_gemm_kernel<4><<<dim3(1, L_SEQ / 64), 128, 0, stream>>>(hbf, wbas, ba, L_SEQ, 64, H_DIM);

    // conv + activations + gates
    conv_qk_kernel<<<dim3(L_SEQ, 32), 128, 0, stream>>>(proj, conv_w, qbuf, kbuf);
    conv_v_kernel<<<(L_SEQ * VALUE_DIM + 255) / 256, 256, 0, stream>>>(proj, conv_w, vbuf);
    gates_kernel<<<(L_SEQ * HV + 255) / 256, 256, 0, stream>>>(ba, dt_bias, A_log, gbuf, betabuf);

    // sequential recurrence, 256-way parallel (32 heads x 8 dv-chunks)
    scan_kernel<<<dim3(8, HV), 128, 0, stream>>>(qbuf, kbuf, vbuf, gbuf, betabuf, corebuf);

    // norm/gate -> bf16, then final WMMA GEMM into d_out
    gate_norm_kernel<<<dim3(L_SEQ, HV), 128, 0, stream>>>(corebuf, proj, norm_w, corebf);
    wmma_gemm_kernel<8><<<dim3(H_DIM / 128, L_SEQ / 64), 128, 0, stream>>>(corebf, wouts, out, L_SEQ, H_DIM, VALUE_DIM);
}
